// Attention_sample_21225728377273
// MI455X (gfx1250) — matrix operand order repaired and checked by reading, not compiled, not run on hardware
//
#include <hip/hip_runtime.h>
#include <hip/hip_bf16.h>
#include <math.h>

#define NHEADS 8
#define KPTS   9
#define BATCH  2
#define HH     128
#define WW     128
#define NPIX   (HH*WW)          // 16384
#define CDIM   256
#define HD     32
#define M_TOT  (BATCH*NPIX)     // 32768
#define QKV_N  (3*CDIM)         // 768

typedef __attribute__((ext_vector_type(16))) __bf16 v16bf;
typedef __attribute__((ext_vector_type(8)))  __bf16 v8bf;
typedef __attribute__((ext_vector_type(8)))  float  v8f;

__device__ __forceinline__ unsigned short f32_to_bf16_rne(float f) {
    unsigned int u = __float_as_uint(f);
    unsigned int r = u + 0x7FFFu + ((u >> 16) & 1u);
    return (unsigned short)(r >> 16);
}

// ---------------------------------------------------------------------------
// fp32 -> bf16 cast (RNE)
// ---------------------------------------------------------------------------
__global__ void cast_f32_bf16_kernel(const float* __restrict__ src,
                                     unsigned short* __restrict__ dst, int n) {
    int i = blockIdx.x * blockDim.x + threadIdx.x;
    if (i < n) dst[i] = f32_to_bf16_rne(src[i]);
}

// ---------------------------------------------------------------------------
// C(MxNcols) = A(Mx256) * W^T  (+bias), W is (Ncols x 256) row-major.
// bf16 inputs, fp32 accumulate via v_wmma_f32_16x16x32_bf16.
// Wave tile: 16 rows x 64 cols. Block: 8 waves -> 128 rows x 64 cols.
// K loop is double-buffered: fragments for step k+1 are in flight while the
// WMMAs for step k execute, so waits become partial instead of loadcnt==0.
// grid = (M/128, Ncols/64)
// ---------------------------------------------------------------------------
__device__ __forceinline__ void load_frags(const unsigned short* __restrict__ arow,
                                           const unsigned short* __restrict__ Wt,
                                           int nbase, int l15, int half, int kb,
                                           v16bf& a, v16bf& b0, v16bf& b1,
                                           v16bf& b2, v16bf& b3) {
    // A fragment layout (16x32 bf16, wave32):
    //   16 contiguous K at kb+half*16
    a = *(const v16bf*)(arow + kb + half * 16);
    // B fragment: col N = nbase + 16*t + l15, 16 contiguous K at kb+half*16
    const int kk = kb + half * 16;
    b0 = *(const v16bf*)(Wt + (size_t)(nbase +  0 + l15) * 256 + kk);
    b1 = *(const v16bf*)(Wt + (size_t)(nbase + 16 + l15) * 256 + kk);
    b2 = *(const v16bf*)(Wt + (size_t)(nbase + 32 + l15) * 256 + kk);
    b3 = *(const v16bf*)(Wt + (size_t)(nbase + 48 + l15) * 256 + kk);
}

__global__ void wmma_gemm_bf16_kernel(const unsigned short* __restrict__ A,
                                      const unsigned short* __restrict__ Wt,
                                      const float* __restrict__ bias,
                                      float* __restrict__ C, int Ncols) {
    const int lane  = threadIdx.x & 31;
    const int wave  = threadIdx.x >> 5;              // 0..7
    const int mbase = (blockIdx.x * 8 + wave) * 16;
    const int nbase = blockIdx.y * 64;
    const int half  = lane >> 4;                     // 0 / 1
    const int l15   = lane & 15;

    v8f acc0 = {}, acc1 = {}, acc2 = {}, acc3 = {};

    const unsigned short* arow = A + (size_t)(mbase + l15) * 256;

    // Ping-pong fragment registers.
    v16bf a0, b00, b01, b02, b03;
    v16bf a1, b10, b11, b12, b13;

    load_frags(arow, Wt, nbase, l15, half, 0, a0, b00, b01, b02, b03);

#pragma unroll
    for (int kb = 0; kb < 256; kb += 64) {
        // Prefetch odd step while even step computes.
        load_frags(arow, Wt, nbase, l15, half, kb + 32, a1, b10, b11, b12, b13);

        acc0 = __builtin_amdgcn_wmma_f32_16x16x32_bf16(false, a0, false, b00, (short)0, acc0, false, false);
        acc1 = __builtin_amdgcn_wmma_f32_16x16x32_bf16(false, a0, false, b01, (short)0, acc1, false, false);
        acc2 = __builtin_amdgcn_wmma_f32_16x16x32_bf16(false, a0, false, b02, (short)0, acc2, false, false);
        acc3 = __builtin_amdgcn_wmma_f32_16x16x32_bf16(false, a0, false, b03, (short)0, acc3, false, false);

        // Prefetch next even step while odd step computes.
        if (kb + 64 < 256)
            load_frags(arow, Wt, nbase, l15, half, kb + 64, a0, b00, b01, b02, b03);

        acc0 = __builtin_amdgcn_wmma_f32_16x16x32_bf16(false, a1, false, b10, (short)0, acc0, false, false);
        acc1 = __builtin_amdgcn_wmma_f32_16x16x32_bf16(false, a1, false, b11, (short)0, acc1, false, false);
        acc2 = __builtin_amdgcn_wmma_f32_16x16x32_bf16(false, a1, false, b12, (short)0, acc2, false, false);
        acc3 = __builtin_amdgcn_wmma_f32_16x16x32_bf16(false, a1, false, b13, (short)0, acc3, false, false);
    }

    float bv0 = bias ? bias[nbase +  0 + l15] : 0.0f;
    float bv1 = bias ? bias[nbase + 16 + l15] : 0.0f;
    float bv2 = bias ? bias[nbase + 32 + l15] : 0.0f;
    float bv3 = bias ? bias[nbase + 48 + l15] : 0.0f;

#pragma unroll
    for (int r = 0; r < 8; ++r) {
        int row = mbase + r + half * 8;
        float* crow = C + (size_t)row * Ncols + nbase + l15;
        crow[0]  = acc0[r] + bv0;
        crow[16] = acc1[r] + bv1;
        crow[32] = acc2[r] + bv2;
        crow[48] = acc3[r] + bv3;
    }
}

// ---------------------------------------------------------------------------
// Per-head offset projection: off[(b*N+n)*8+h][j] = sum_c Woff[j][c]*x[b,n,h*32+c]
// ---------------------------------------------------------------------------
__global__ void offset_kernel(const float* __restrict__ x,
                              const float* __restrict__ Woff,
                              float* __restrict__ off) {
    __shared__ float w[2 * KPTS * HD];
    for (int i = threadIdx.x; i < 2 * KPTS * HD; i += blockDim.x) w[i] = Woff[i];
    __syncthreads();

    int t = blockIdx.x * blockDim.x + threadIdx.x;
    if (t >= M_TOT * NHEADS * 2 * KPTS) return;
    int j  = t % 18;
    int h  = (t / 18) & 7;
    int mn = t / (18 * 8);                  // b*N + n
    const float* xr = x + (size_t)mn * CDIM + h * HD;
    float s = 0.f;
#pragma unroll
    for (int c = 0; c < HD; ++c) s += w[j * HD + c] * xr[c];
    off[t] = s;
}

// ---------------------------------------------------------------------------
// Deformable sampling + 9-way softmax attention. One thread per (group,pixel).
// k/v are read channel-last directly out of the qkv buffer (vec4 taps).
// Output written bf16 channel-last (B,N,C) ready for the proj WMMA GEMM.
// ---------------------------------------------------------------------------
struct Taps { int i00, i01, i10, i11; float w00, w01, w10, w11; };

__device__ __forceinline__ Taps make_taps(float sy, float sx) {
    float y0f = floorf(sy), x0f = floorf(sx);
    int y0 = (int)y0f, x0 = (int)x0f;
    float wy1 = sy - y0f, wx1 = sx - x0f;
    float wy0 = 1.f - wy1, wx0 = 1.f - wx1;
    int y1 = y0 + 1, x1 = x0 + 1;
    bool vy0 = (y0 >= 0) && (y0 < HH), vy1 = (y1 >= 0) && (y1 < HH);
    bool vx0 = (x0 >= 0) && (x0 < WW), vx1 = (x1 >= 0) && (x1 < WW);
    Taps tp;
    tp.w00 = (vy0 && vx0) ? wy0 * wx0 : 0.f;
    tp.w01 = (vy0 && vx1) ? wy0 * wx1 : 0.f;
    tp.w10 = (vy1 && vx0) ? wy1 * wx0 : 0.f;
    tp.w11 = (vy1 && vx1) ? wy1 * wx1 : 0.f;
    int cy0 = min(max(y0, 0), HH - 1), cy1 = min(max(y1, 0), HH - 1);
    int cx0 = min(max(x0, 0), WW - 1), cx1 = min(max(x1, 0), WW - 1);
    tp.i00 = cy0 * WW + cx0; tp.i01 = cy0 * WW + cx1;
    tp.i10 = cy1 * WW + cx0; tp.i11 = cy1 * WW + cx1;
    return tp;
}

__global__ void deform_attn_kernel(const float* __restrict__ qkv,
                                   const float* __restrict__ off,
                                   unsigned short* __restrict__ attn_out) {
    int t = blockIdx.x * blockDim.x + threadIdx.x;
    if (t >= BATCH * NHEADS * NPIX) return;
    int n = t % NPIX;
    int g = t / NPIX;
    int h = g & 7;
    int b = g >> 3;
    int py = n / WW, px = n % WW;

    const float* qrow = qkv + (size_t)(b * NPIX + n) * QKV_N + h * HD;
    float q[HD];
#pragma unroll
    for (int c = 0; c < HD; c += 4) {
        float4 qq = *(const float4*)(qrow + c);
        q[c] = qq.x; q[c + 1] = qq.y; q[c + 2] = qq.z; q[c + 3] = qq.w;
    }

    const float* offp = off + ((size_t)(b * NPIX + n) * NHEADS + h) * (2 * KPTS);
    float sy_a[KPTS], sx_a[KPTS];
#pragma unroll
    for (int p = 0; p < KPTS; ++p) {
        sy_a[p] = (float)py + (float)(p / 3 - 1) + offp[2 * p];
        sx_a[p] = (float)px + (float)(p % 3 - 1) + offp[2 * p + 1];
    }

    const size_t kbase = (size_t)b * NPIX * QKV_N + CDIM + h * HD;      // k channel-last
    const size_t vbase = kbase + CDIM;                                   // v channel-last

    float logits[KPTS];
#pragma unroll
    for (int p = 0; p < KPTS; ++p) {
        Taps tp = make_taps(sy_a[p], sx_a[p]);
        const float* k00 = qkv + kbase + (size_t)tp.i00 * QKV_N;
        const float* k01 = qkv + kbase + (size_t)tp.i01 * QKV_N;
        const float* k10 = qkv + kbase + (size_t)tp.i10 * QKV_N;
        const float* k11 = qkv + kbase + (size_t)tp.i11 * QKV_N;
        float acc = 0.f;
#pragma unroll
        for (int c = 0; c < HD; c += 4) {
            float4 a = *(const float4*)(k00 + c);
            float4 bq = *(const float4*)(k01 + c);
            float4 cq = *(const float4*)(k10 + c);
            float4 dq = *(const float4*)(k11 + c);
            float s0 = tp.w00 * a.x + tp.w01 * bq.x + tp.w10 * cq.x + tp.w11 * dq.x;
            float s1 = tp.w00 * a.y + tp.w01 * bq.y + tp.w10 * cq.y + tp.w11 * dq.y;
            float s2 = tp.w00 * a.z + tp.w01 * bq.z + tp.w10 * cq.z + tp.w11 * dq.z;
            float s3 = tp.w00 * a.w + tp.w01 * bq.w + tp.w10 * cq.w + tp.w11 * dq.w;
            acc += q[c] * s0 + q[c + 1] * s1 + q[c + 2] * s2 + q[c + 3] * s3;
        }
        logits[p] = acc * 0.17677669529663687f;   // hd^-0.5
    }

    float mx = logits[0];
#pragma unroll
    for (int p = 1; p < KPTS; ++p) mx = fmaxf(mx, logits[p]);
    float wgt[KPTS], sum = 0.f;
#pragma unroll
    for (int p = 0; p < KPTS; ++p) { wgt[p] = __expf(logits[p] - mx); sum += wgt[p]; }
    float inv = 1.f / sum;

    float out[HD];
#pragma unroll
    for (int c = 0; c < HD; ++c) out[c] = 0.f;

#pragma unroll
    for (int p = 0; p < KPTS; ++p) {
        Taps tp = make_taps(sy_a[p], sx_a[p]);
        float ap  = wgt[p] * inv;
        float c00 = ap * tp.w00, c01 = ap * tp.w01, c10 = ap * tp.w10, c11 = ap * tp.w11;
        const float* v00 = qkv + vbase + (size_t)tp.i00 * QKV_N;
        const float* v01 = qkv + vbase + (size_t)tp.i01 * QKV_N;
        const float* v10 = qkv + vbase + (size_t)tp.i10 * QKV_N;
        const float* v11 = qkv + vbase + (size_t)tp.i11 * QKV_N;
#pragma unroll
        for (int c = 0; c < HD; c += 4) {
            float4 a = *(const float4*)(v00 + c);
            float4 bq = *(const float4*)(v01 + c);
            float4 cq = *(const float4*)(v10 + c);
            float4 dq = *(const float4*)(v11 + c);
            out[c]     += c00 * a.x + c01 * bq.x + c10 * cq.x + c11 * dq.x;
            out[c + 1] += c00 * a.y + c01 * bq.y + c10 * cq.y + c11 * dq.y;
            out[c + 2] += c00 * a.z + c01 * bq.z + c10 * cq.z + c11 * dq.z;
            out[c + 3] += c00 * a.w + c01 * bq.w + c10 * cq.w + c11 * dq.w;
        }
    }

    unsigned short* orow = attn_out + (size_t)(b * NPIX + n) * CDIM + h * HD;
#pragma unroll
    for (int c = 0; c < HD; ++c) orow[c] = f32_to_bf16_rne(out[c]);
}

// ---------------------------------------------------------------------------
// Launcher
// ---------------------------------------------------------------------------
extern "C" void kernel_launch(void* const* d_in, const int* in_sizes, int n_in,
                              void* d_out, int out_size, void* d_ws, size_t ws_size,
                              hipStream_t stream) {
    const float* x     = (const float*)d_in[0];   // (B,N,C)
    const float* Wqkv  = (const float*)d_in[1];   // (3C, C)
    const float* Woff  = (const float*)d_in[2];   // (18, hd)
    const float* Wproj = (const float*)d_in[3];   // (C, C)
    const float* bproj = (const float*)d_in[4];   // (C,)
    // d_in[5], d_in[6] = H, W scalars (compile-time constants here)

    char* ws = (char*)d_ws;
    const size_t SZ_XBF   = (size_t)M_TOT * CDIM * 2;           // 16 MB
    const size_t SZ_WQKV  = (size_t)QKV_N * CDIM * 2;           // 384 KB
    const size_t SZ_WPROJ = (size_t)CDIM * CDIM * 2;            // 128 KB
    const size_t SZ_QKV   = (size_t)M_TOT * QKV_N * 4;          // 96 MB
    const size_t SZ_OFF   = (size_t)M_TOT * NHEADS * 2 * KPTS * 4; // 18.9 MB

    unsigned short* x_bf     = (unsigned short*)(ws);
    unsigned short* wqkv_bf  = (unsigned short*)(ws + SZ_XBF);
    unsigned short* wproj_bf = (unsigned short*)(ws + SZ_XBF + SZ_WQKV);
    float*          qkv      = (float*)(ws + SZ_XBF + SZ_WQKV + SZ_WPROJ);
    float*          offb     = (float*)(ws + SZ_XBF + SZ_WQKV + SZ_WPROJ + SZ_QKV);
    unsigned short* attn_bf  = (unsigned short*)(ws + SZ_XBF + SZ_WQKV + SZ_WPROJ + SZ_QKV + SZ_OFF);

    // 1) casts to bf16
    {
        int n = M_TOT * CDIM;
        cast_f32_bf16_kernel<<<(n + 255) / 256, 256, 0, stream>>>(x, x_bf, n);
        n = QKV_N * CDIM;
        cast_f32_bf16_kernel<<<(n + 255) / 256, 256, 0, stream>>>(Wqkv, wqkv_bf, n);
        n = CDIM * CDIM;
        cast_f32_bf16_kernel<<<(n + 255) / 256, 256, 0, stream>>>(Wproj, wproj_bf, n);
    }

    // 2) QKV projection: (32768 x 256) * (768 x 256)^T -> (32768 x 768) fp32
    {
        dim3 grid(M_TOT / 128, QKV_N / 64);
        wmma_gemm_bf16_kernel<<<grid, 256, 0, stream>>>(x_bf, wqkv_bf, nullptr, qkv, QKV_N);
    }

    // 3) per-head offsets
    {
        int n = M_TOT * NHEADS * 2 * KPTS;
        offset_kernel<<<(n + 255) / 256, 256, 0, stream>>>(x, Woff, offb);
    }

    // 4) deformable sampling + softmax attention -> bf16 (B,N,C)
    {
        int n = BATCH * NHEADS * NPIX;
        deform_attn_kernel<<<(n + 255) / 256, 256, 0, stream>>>(qkv, offb, attn_bf);
    }

    // 5) output projection: (32768 x 256) * (256 x 256)^T + b -> d_out fp32
    {
        dim3 grid(M_TOT / 128, CDIM / 64);
        wmma_gemm_bf16_kernel<<<grid, 256, 0, stream>>>(attn_bf, wproj_bf, bproj,
                                                        (float*)d_out, CDIM);
    }
}
